// SwinBlock_80479097192711
// MI455X (gfx1250) — matrix operand order repaired and checked by reading, not compiled, not run on hardware
//
#include <hip/hip_runtime.h>

// ---------------------------------------------------------------------------
// Swin block for MI455X (gfx1250, wave32, WMMA).
// B=2, S=8192, D=1024, NW=16 (window n=512), SHIFT=256, NH=16, DH=64, DFF=4096
// All matrix math runs through v_wmma_f32_16x16x32_f16 (f16 in, f32 accum).
// Row-major LDS staging uses GLOBAL_LOAD_ASYNC_TO_LDS_B128 when the toolchain
// exposes the builtin (ASYNCcnt-tracked CDNA5 async copy), else load+ds_store.
// ---------------------------------------------------------------------------

typedef _Float16 f16;
typedef __attribute__((ext_vector_type(16))) _Float16 v16h;
typedef __attribute__((ext_vector_type(8)))  float    v8f;
typedef int v4i __attribute__((vector_size(16)));

union HFrag { v16h v; uint4 q[2]; };   // 16 f16 = two 16B chunks
union Frag8 { uint4 u; f16 h[8]; };

#if __has_builtin(__builtin_amdgcn_global_load_async_to_lds_b128) && \
    __has_builtin(__builtin_amdgcn_s_wait_asynccnt)
#define USE_ASYNC_COPY 1
#endif

#ifdef USE_ASYNC_COPY
typedef __attribute__((address_space(1))) v4i* global_v4i_p;
typedef __attribute__((address_space(3))) v4i* lds_v4i_p;
// Generic->LDS addrspace(3): the LDS address is the low 32 bits of the generic pointer.
__device__ __forceinline__ void async_copy_b128(const void* g, void* l) {
  __builtin_amdgcn_global_load_async_to_lds_b128(
      (global_v4i_p)(uintptr_t)g,
      (lds_v4i_p)(uint32_t)(uintptr_t)l,
      0, 0);
}
__device__ __forceinline__ void async_copy_wait() {
  __builtin_amdgcn_s_wait_asynccnt(0);
}
#endif

__device__ __forceinline__ float gelu_f(float x) {
  const float k = 0.7978845608028654f;                 // sqrt(2/pi)
  float t = tanhf(k * (x + 0.044715f * x * x * x));
  return 0.5f * x * (1.0f + t);
}

// ---------------------------------------------------------------------------
// f32 -> f16 cast (weights)
// ---------------------------------------------------------------------------
__global__ __launch_bounds__(256)
void cast_f32_to_f16(const float* __restrict__ in, f16* __restrict__ out, int n) {
  int i = blockIdx.x * 256 + threadIdx.x;
  const int stride = gridDim.x * 256;
  for (; i < n; i += stride) out[i] = (f16)in[i];
}

// ---------------------------------------------------------------------------
// RMSNorm (+ optional token roll fused into the read index), f16 output.
// One block per token row; 256 threads * float4 = 1024 elems.
// ---------------------------------------------------------------------------
template<int SHIFT_AMT>
__global__ __launch_bounds__(256)
void rmsnorm_shift(const float* __restrict__ x, const float* __restrict__ g,
                   f16* __restrict__ out) {
  const int row = blockIdx.x;                 // 0 .. B*S-1
  const int b = row >> 13;
  const int t = row & 8191;
  const int src = (b << 13) | ((t + SHIFT_AMT) & 8191);
  const float4 v = ((const float4*)(x + (size_t)src * 1024))[threadIdx.x];
  float ss = v.x * v.x + v.y * v.y + v.z * v.z + v.w * v.w;
#pragma unroll
  for (int m = 16; m >= 1; m >>= 1) ss += __shfl_xor(ss, m, 32);
  __shared__ float wsum[8];
  if ((threadIdx.x & 31) == 0) wsum[threadIdx.x >> 5] = ss;
  __syncthreads();
  float tot = 0.f;
#pragma unroll
  for (int i = 0; i < 8; ++i) tot += wsum[i];
  const float rinv = rsqrtf(tot * (1.0f / 1024.0f) + 1e-6f);
  const int c = threadIdx.x * 4;
  const float4 gg = *(const float4*)(g + c);
  union { f16 h[4]; uint2 u; } ov;
  ov.h[0] = (f16)(v.x * gg.x * rinv);
  ov.h[1] = (f16)(v.y * gg.y * rinv);
  ov.h[2] = (f16)(v.z * gg.z * rinv);
  ov.h[3] = (f16)(v.w * gg.w * rinv);
  *(uint2*)(out + (size_t)row * 1024 + c) = ov.u;
}

// ---------------------------------------------------------------------------
// Tiled WMMA GEMM: C[M,N] = A[M,K] (f16, row-major) * B[K,N] (f16, row-major)
// Workgroup tile 128x256, BK=32, 8 waves (2x4), each wave 64x64 (4x4 WMMA
// tiles) -> 16 WMMAs per K-step against 16 ds_load_b128 (4x fragment reuse).
// EPI: 0 = QKV head-major f16 store
//      1 = o-proj: roll(-256) + residual x, f32 store
//      2 = FFN1: bias + gelu, f16 store
//      3 = FFN2: bias + residual, f32 store
// ---------------------------------------------------------------------------
template<int EPI>
__global__ __launch_bounds__(256)
void gemm_f16(const f16* __restrict__ A, const f16* __restrict__ Bw,
              int M, int N, int K,
              f16* __restrict__ out16, float* __restrict__ out32,
              const float* __restrict__ bias, const float* __restrict__ resid) {
  __shared__ f16 As[128 * 32];        // [row][k]           8 KB
  __shared__ f16 Bs[256 * 32];        // [col][k] transposed 16 KB

  const int tid = threadIdx.x, lane = tid & 31, wid = tid >> 5;
  const int m0 = blockIdx.y * 128, n0 = blockIdx.x * 256;
  const int wr = (wid & 1) * 64;      // wave row offset in tile
  const int wc = (wid >> 1) * 64;     // wave col offset in tile

  v8f acc[4][4] = {};

  const int lr = lane & 15;
  const int ah = (lane < 16) ? 0 : 8;     // A K-half start
  const int bh = (lane < 16) ? 0 : 8;     // B K-half start

  for (int kb = 0; kb < K; kb += 32) {
    __syncthreads();
    // ---- stage A tile 128x32 (row-major; async path if available) ----
#pragma unroll
    for (int i = 0; i < 2; ++i) {
      const int ch = tid * 2 + i;
      const int r = ch >> 2, kc = (ch & 3) * 8;
#ifdef USE_ASYNC_COPY
      async_copy_b128(A + (size_t)(m0 + r) * K + kb + kc, As + r * 32 + kc);
#else
      *(uint4*)(As + r * 32 + kc) =
          *(const uint4*)(A + (size_t)(m0 + r) * K + kb + kc);
#endif
    }
    // ---- stage B tile 32x256, transposed to [col][k] ----
#pragma unroll
    for (int i = 0; i < 4; ++i) {
      const int ch = tid * 4 + i;
      const int k = ch >> 5, nc = (ch & 31) * 8;
      Frag8 bv; bv.u = *(const uint4*)(Bw + (size_t)(kb + k) * N + n0 + nc);
#pragma unroll
      for (int e = 0; e < 8; ++e) Bs[(nc + e) * 32 + k] = bv.h[e];
    }
#ifdef USE_ASYNC_COPY
    async_copy_wait();
#endif
    __syncthreads();

    HFrag af[4], bf[4];
#pragma unroll
    for (int rt = 0; rt < 4; ++rt) {
      const f16* ap = As + (wr + rt * 16 + lr) * 32 + ah;
      af[rt].q[0] = *(const uint4*)ap;
      af[rt].q[1] = *(const uint4*)(ap + 16);
    }
#pragma unroll
    for (int ct = 0; ct < 4; ++ct) {
      const f16* bp = Bs + (wc + ct * 16 + lr) * 32 + bh;
      bf[ct].q[0] = *(const uint4*)bp;
      bf[ct].q[1] = *(const uint4*)(bp + 16);
    }
#pragma unroll
    for (int rt = 0; rt < 4; ++rt)
#pragma unroll
      for (int ct = 0; ct < 4; ++ct)
        acc[rt][ct] = __builtin_amdgcn_wmma_f32_16x16x32_f16(
            false, af[rt].v, false, bf[ct].v, (short)0, acc[rt][ct], false, false);
  }

  // ----- epilogue -----
  const int rsel = (lane >> 4) << 3;        // 0 or 8
#pragma unroll
  for (int rt = 0; rt < 4; ++rt)
#pragma unroll
    for (int ct = 0; ct < 4; ++ct)
#pragma unroll
      for (int j = 0; j < 8; ++j) {
        const int r = m0 + wr + rt * 16 + j + rsel;
        const int c = n0 + wc + ct * 16 + lr;
        const float val = acc[rt][ct][j];
        if constexpr (EPI == 0) {           // QKV -> [b,w,h,t,d] f16
          const int b = r >> 13, ts = r & 8191, w = ts >> 9, t = ts & 511;
          const int h = c >> 6, d = c & 63;
          const size_t idx =
              ((size_t)(((b * 16 + w) * 16 + h) * 512 + t)) * 64 + d;
          out16[idx] = (f16)val;
        } else if constexpr (EPI == 1) {    // o-proj: roll + residual, f32
          const int b = r >> 13, ts = r & 8191;
          const int dt = (ts + 8192 - 256) & 8191;
          const size_t dr = ((size_t)b << 13) + dt;
          out32[dr * (size_t)N + c] = val + resid[dr * (size_t)N + c];
        } else if constexpr (EPI == 2) {    // FFN1: bias + gelu, f16
          out16[(size_t)r * N + c] = (f16)gelu_f(val + bias[c]);
        } else {                            // FFN2: bias + residual, f32
          out32[(size_t)r * N + c] =
              val + bias[c] + resid[(size_t)r * N + c];
        }
      }
}

// ---------------------------------------------------------------------------
// Windowed flash attention. One block per (b, window, head) = 512 blocks.
// 8 waves; each wave owns 64 q rows (4 tiles of 16). K kept row-major in LDS
// (async-staged when available), V transposed in LDS so both WMMA B-operands
// are contiguous 16B reads. Online softmax over 32-key chunks; P goes
// C-layout -> A-layout through a per-wave LDS staging tile (in-order DS).
// ---------------------------------------------------------------------------
__global__ __launch_bounds__(256)
void attn_window(const f16* __restrict__ Q, const f16* __restrict__ Km,
                 const f16* __restrict__ Vm, f16* __restrict__ O) {
  extern __shared__ char smem_raw[];
  f16* Ks = (f16*)smem_raw;          // [512][64]
  f16* Vt = Ks + 512 * 64;           // [64][512]
  f16* Ps = Vt + 64 * 512;           // [8 waves][16][32]

  const int hw = blockIdx.x;         // ((b*16+w)*16 + h)
  const int tid = threadIdx.x, lane = tid & 31, wid = tid >> 5;
  const size_t base = (size_t)hw * (512 * 64);

  for (int ch = tid; ch < 4096; ch += 256) {  // stage K and V^T (64KB each)
    const int n = ch >> 3, d = (ch & 7) << 3;
#ifdef USE_ASYNC_COPY
    async_copy_b128(Km + base + n * 64 + d, Ks + n * 64 + d);
#else
    *(uint4*)(Ks + n * 64 + d) = *(const uint4*)(Km + base + n * 64 + d);
#endif
    Frag8 vv; vv.u = *(const uint4*)(Vm + base + n * 64 + d);
#pragma unroll
    for (int e = 0; e < 8; ++e) Vt[(d + e) * 512 + n] = vv.h[e];
  }
#ifdef USE_ASYNC_COPY
  async_copy_wait();
#endif
  __syncthreads();

  const int lr = lane & 15;
  const int rsel = (lane >> 4) << 3;        // 0 or 8 (C-layout row select)
  const int ah = (lane < 16) ? 0 : 8;       // A operand K-half
  const int bh = (lane < 16) ? 0 : 8;       // B operand K-half

  const int b = hw >> 8, w = (hw >> 4) & 15, h = hw & 15;
  const size_t orow0 = ((size_t)b << 13) + (size_t)w * 512;

  for (int qt = 0; qt < 4; ++qt) {
    const int qbase = wid * 64 + qt * 16;
    const f16* qp = Q + base + (size_t)(qbase + lr) * 64;
    HFrag qf[2];
    qf[0].q[0] = *(const uint4*)(qp + ah);
    qf[0].q[1] = *(const uint4*)(qp + ah + 16);
    qf[1].q[0] = *(const uint4*)(qp + 32 + ah);
    qf[1].q[1] = *(const uint4*)(qp + 32 + ah + 16);

    v8f oacc[4] = {};
    float mrow[8], lrow[8];
#pragma unroll
    for (int j = 0; j < 8; ++j) { mrow[j] = -3.0e38f; lrow[j] = 0.f; }

    for (int kc = 0; kc < 16; ++kc) {       // 32 keys per chunk
      const int kb = kc * 32;
      v8f s0 = {}, s1 = {};
#pragma unroll
      for (int ds = 0; ds < 2; ++ds) {      // accumulate over d (DH=64)
        const f16* kp0 = Ks + (size_t)(kb + lr) * 64 + ds * 32 + bh;
        HFrag b0; b0.q[0] = *(const uint4*)kp0; b0.q[1] = *(const uint4*)(kp0 + 16);
        s0 = __builtin_amdgcn_wmma_f32_16x16x32_f16(
            false, qf[ds].v, false, b0.v, (short)0, s0, false, false);
        const f16* kp1 = Ks + (size_t)(kb + 16 + lr) * 64 + ds * 32 + bh;
        HFrag b1; b1.q[0] = *(const uint4*)kp1; b1.q[1] = *(const uint4*)(kp1 + 16);
        s1 = __builtin_amdgcn_wmma_f32_16x16x32_f16(
            false, qf[ds].v, false, b1.v, (short)0, s1, false, false);
      }
      // online softmax over these 32 keys
      f16* pw = Ps + wid * 512;
#pragma unroll
      for (int j = 0; j < 8; ++j) {
        const float a0 = s0[j] * 0.125f, a1 = s1[j] * 0.125f;   // 1/sqrt(64)
        float mx = fmaxf(a0, a1);
#pragma unroll
        for (int msk = 8; msk >= 1; msk >>= 1)
          mx = fmaxf(mx, __shfl_xor(mx, msk, 32));
        const float nm = fmaxf(mrow[j], mx);
        const float corr = __expf(mrow[j] - nm);
        const float p0 = __expf(a0 - nm), p1 = __expf(a1 - nm);
        float rs = p0 + p1;
#pragma unroll
        for (int msk = 8; msk >= 1; msk >>= 1) rs += __shfl_xor(rs, msk, 32);
        lrow[j] = lrow[j] * corr + rs;
        mrow[j] = nm;
        oacc[0][j] *= corr; oacc[1][j] *= corr;
        oacc[2][j] *= corr; oacc[3][j] *= corr;
        f16* pr = pw + (j + rsel) * 32;     // P staged row-major 16x32
        pr[lr] = (f16)p0;
        pr[16 + lr] = (f16)p1;
      }
      // reload P in A-operand layout (same wave; LDS is in-order)
      HFrag pf;
      const f16* pp = pw + lr * 32 + ah;
      pf.q[0] = *(const uint4*)pp;
      pf.q[1] = *(const uint4*)(pp + 16);
#pragma unroll
      for (int t = 0; t < 4; ++t) {
        const f16* vp = Vt + (size_t)(t * 16 + lr) * 512 + kb + bh;
        HFrag vf; vf.q[0] = *(const uint4*)vp; vf.q[1] = *(const uint4*)(vp + 16);
        oacc[t] = __builtin_amdgcn_wmma_f32_16x16x32_f16(
            false, pf.v, false, vf.v, (short)0, oacc[t], false, false);
      }
    }
    // normalize + store to [token, h*64+d] f16
#pragma unroll
    for (int j = 0; j < 8; ++j) {
      const float inv = 1.0f / lrow[j];
      const int qrow = qbase + j + rsel;
      const size_t rowp = (orow0 + qrow) * 1024 + (size_t)h * 64;
      O[rowp + 0 + lr]  = (f16)(oacc[0][j] * inv);
      O[rowp + 16 + lr] = (f16)(oacc[1][j] * inv);
      O[rowp + 32 + lr] = (f16)(oacc[2][j] * inv);
      O[rowp + 48 + lr] = (f16)(oacc[3][j] * inv);
    }
  }
}

// ---------------------------------------------------------------------------
// Host launcher
// ---------------------------------------------------------------------------
extern "C" void kernel_launch(void* const* d_in, const int* in_sizes, int n_in,
                              void* d_out, int out_size, void* d_ws, size_t ws_size,
                              hipStream_t stream) {
  (void)in_sizes; (void)n_in; (void)out_size; (void)ws_size;
  const float* x      = (const float*)d_in[0];
  const float* wq     = (const float*)d_in[1];
  const float* wk     = (const float*)d_in[2];
  const float* wv     = (const float*)d_in[3];
  const float* wo     = (const float*)d_in[4];
  const float* g_attn = (const float*)d_in[5];
  const float* g_ff   = (const float*)d_in[6];
  const float* w1     = (const float*)d_in[7];
  const float* b1     = (const float*)d_in[8];
  const float* w2     = (const float*)d_in[9];
  const float* b2     = (const float*)d_in[10];
  float* out = (float*)d_out;

  const int MD = 16384;        // B*S
  const int Dm = 1024, DFF = 4096;

  char* ws = (char*)d_ws;
  size_t off = 0;
  auto alloc = [&](size_t bytes) -> void* {
    void* p = ws + off;
    off += (bytes + 255) & ~(size_t)255;
    return p;
  };
  f16* wq16   = (f16*)alloc((size_t)Dm * Dm * 2);
  f16* wk16   = (f16*)alloc((size_t)Dm * Dm * 2);
  f16* wv16   = (f16*)alloc((size_t)Dm * Dm * 2);
  f16* wo16   = (f16*)alloc((size_t)Dm * Dm * 2);
  f16* w116   = (f16*)alloc((size_t)Dm * DFF * 2);
  f16* w216   = (f16*)alloc((size_t)DFF * Dm * 2);
  f16* hshift = (f16*)alloc((size_t)MD * Dm * 2);
  f16* q16    = (f16*)alloc((size_t)MD * Dm * 2);
  f16* k16    = (f16*)alloc((size_t)MD * Dm * 2);
  f16* v16    = (f16*)alloc((size_t)MD * Dm * 2);
  f16* o16    = (f16*)alloc((size_t)MD * Dm * 2);
  float* x1   = (float*)alloc((size_t)MD * Dm * 4);
  f16* h2     = (f16*)alloc((size_t)MD * Dm * 2);
  f16* ff1    = (f16*)alloc((size_t)MD * DFF * 2);

  // 1) weight casts
  cast_f32_to_f16<<<2048, 256, 0, stream>>>(wq, wq16, Dm * Dm);
  cast_f32_to_f16<<<2048, 256, 0, stream>>>(wk, wk16, Dm * Dm);
  cast_f32_to_f16<<<2048, 256, 0, stream>>>(wv, wv16, Dm * Dm);
  cast_f32_to_f16<<<2048, 256, 0, stream>>>(wo, wo16, Dm * Dm);
  cast_f32_to_f16<<<4096, 256, 0, stream>>>(w1, w116, Dm * DFF);
  cast_f32_to_f16<<<4096, 256, 0, stream>>>(w2, w216, DFF * Dm);

  // 2) RMSNorm + roll(-256)
  rmsnorm_shift<256><<<MD, 256, 0, stream>>>(x, g_attn, hshift);

  // 3) QKV projections (head-major outputs), tile 128x256
  gemm_f16<0><<<dim3(4, 128), 256, 0, stream>>>(hshift, wq16, MD, Dm, Dm,
                                                q16, nullptr, nullptr, nullptr);
  gemm_f16<0><<<dim3(4, 128), 256, 0, stream>>>(hshift, wk16, MD, Dm, Dm,
                                                k16, nullptr, nullptr, nullptr);
  gemm_f16<0><<<dim3(4, 128), 256, 0, stream>>>(hshift, wv16, MD, Dm, Dm,
                                                v16, nullptr, nullptr, nullptr);

  // 4) windowed flash attention (512 head-windows), 136KB dynamic LDS
  attn_window<<<512, 256, 139264, stream>>>(q16, k16, v16, o16);

  // 5) output projection + roll(-256) + residual -> x1 (f32)
  gemm_f16<1><<<dim3(4, 128), 256, 0, stream>>>(o16, wo16, MD, Dm, Dm,
                                                nullptr, x1, nullptr, x);

  // 6) RMSNorm 2
  rmsnorm_shift<0><<<MD, 256, 0, stream>>>(x1, g_ff, h2);

  // 7) FFN1: h2 @ w1 + b1 -> gelu (f16)
  gemm_f16<2><<<dim3(16, 128), 256, 0, stream>>>(h2, w116, MD, DFF, Dm,
                                                 ff1, nullptr, b1, nullptr);

  // 8) FFN2: ff1 @ w2 + b2 + x1 -> d_out (f32)
  gemm_f16<3><<<dim3(4, 128), 256, 0, stream>>>(ff1, w216, MD, Dm, DFF,
                                                nullptr, out, b2, x1);
}
